// ScaledDotProductAttention_80668075753736
// MI455X (gfx1250) — matrix operand order repaired and checked by reading, not compiled, not run on hardware
//
#include <hip/hip_runtime.h>
#include <hip/hip_bf16.h>

// ---------------------------------------------------------------------------
// Shapes (fixed by the reference): B=4, S=2048, H=16, L=64
// queries/keys/values: [B,S,H,L] f32, mask: [B,1,S,S] i32, out: [B,S,H,L] f32
// ---------------------------------------------------------------------------
#define Bq 4
#define Sq 2048
#define Hh 16
#define Ll 64

typedef __attribute__((ext_vector_type(16))) __bf16 v16bf;
typedef __attribute__((ext_vector_type(8)))  float  v8f;
typedef unsigned int v4u __attribute__((ext_vector_type(4)));
typedef unsigned int v8u __attribute__((ext_vector_type(8)));

union FragBF16 {
    v16bf v;
    unsigned short u[16];
    uint4 q[2];
};

__device__ __forceinline__ unsigned short f32_to_bf16_rne(float f) {
    unsigned int x = __builtin_bit_cast(unsigned int, f);
    x = (x + 0x7FFFu + ((x >> 16) & 1u)) >> 16;   // round-to-nearest-even
    return (unsigned short)x;
}

// LDS byte offset of a __shared__ object (AS3 pointer value == LDS offset).
// Widen before truncating so the (unused) host pass also compiles.
typedef __attribute__((address_space(3))) const void* lds_cptr_t;
__device__ __forceinline__ unsigned lds_offset(const void* p) {
#if defined(__HIP_DEVICE_COMPILE__)
    return (unsigned)(unsigned long long)(lds_cptr_t)p;
#else
    return 0u;
#endif
}

// ---------------------------------------------------------------------------
// TDM: async-load one 16x64 f32 tile (row stride 1024 elements) into LDS.
// D# group0: {ctrl(count=1), lds_addr, global_addr[31:0],
//             global_addr[56:32] | type=2<<30}
// D# group1: data_size=4B, tensor_dim0=64, tensor_dim1=16, tile_dim0=64,
//            tile_dim1=16, tensor_dim0_stride=1024 (data_size units)
// 2-group form (<=2D tensor): tensor_load_to_lds sA, sB. Tracked by TENSORcnt.
// ---------------------------------------------------------------------------
__device__ __forceinline__ void tdm_load_qtile(unsigned lds_addr,
                                               unsigned long long gaddr) {
    v4u g0;
    g0[0] = 1u;                                         // count=1, user mode
    g0[1] = lds_addr;                                   // LDS dest (bytes)
    g0[2] = (unsigned)(gaddr & 0xFFFFFFFFull);          // global addr lo
    g0[3] = (unsigned)((gaddr >> 32) & 0x1FFFFFFull)    // global addr [56:32]
          | (2u << 30);                                 // type = 2 ("image")
    v8u g1;
    g1[0] = (2u << 16);       // workgroup_mask=0 (not in cluster), data_size=4B
    g1[1] = (64u << 16);      // tensor_dim0 = 64 (bits 79:48 low half)
    g1[2] = (16u << 16);      // tensor_dim0 hi=0, tensor_dim1 = 16
    g1[3] = (64u << 16);      // tensor_dim1 hi=0, tile_dim0 = 64
    g1[4] = 16u;              // tile_dim1 = 16, tile_dim2 = 0
    g1[5] = 1024u;            // tensor_dim0_stride = H*L = 1024 elements
    g1[6] = 0u;               // stride hi, tensor_dim1_stride lo
    g1[7] = 0u;
    asm volatile("tensor_load_to_lds %0, %1" :: "s"(g0), "s"(g1) : "memory");
}

// ---------------------------------------------------------------------------
// Kernel 0: zero the 64-entry S[b,h] accumulator in workspace
// ---------------------------------------------------------------------------
__global__ void zero_sbh_kernel(float* __restrict__ p, int n) {
    int i = blockIdx.x * blockDim.x + threadIdx.x;
    if (i < n) p[i] = 0.0f;
}

// ---------------------------------------------------------------------------
// Kernel 1: per (b,h), compute S[b,h] = sum_k ( sum_q p[q,k] ) where
// p = softmax over q of masked (Q K^T / sqrt(L)). Scores via bf16 WMMA;
// Q tiles stream in via TDM (TENSORcnt), double-staged against compute;
// online softmax along q per column; faithful Z/Z normalization.
// Grid: B*H*(S/64) blocks of 128 threads (4 waves; each wave = 16 columns).
// ---------------------------------------------------------------------------
__global__ void attn_colsum_kernel(const float* __restrict__ Q,
                                   const float* __restrict__ K,
                                   const int*   __restrict__ mask,
                                   float*       __restrict__ Sbh) {
    __shared__ __align__(16) float          ldsF[16 * Ll];     // raw f32 TDM dest
    __shared__ __align__(16) unsigned short qtile[16][72];     // bf16 tile, padded

    const int tid  = threadIdx.x;
    const int wave = tid >> 5;
    const int lane = tid & 31;
    const int bx   = blockIdx.x;
    const int kb   = bx & 31;           // 32 strips of 64 columns
    const int h    = (bx >> 5) & 15;
    const int b    = bx >> 9;
    const int k0   = kb * 64 + wave * 16;
    const int n    = lane & 15;         // column within tile (B/C layout)
    const int half = lane >> 4;         // lane half selects K-subrange / row-half

    // --- B fragments: 32x16 bf16 tiles of K rows (constant over q loop) ---
    // B layout: lane -> column n (= K-matrix row k0+n), elems 0-7 -> l = 8*half+i ; 8-15 -> 16+8*half+i-8
    FragBF16 bf0, bf1;
    {
        const float* kp = K + (((size_t)(b * Sq + (k0 + n)) * Hh + h) * Ll);
#pragma unroll
        for (int i = 0; i < 16; ++i) {
            bf0.u[i] = f32_to_bf16_rne(kp[16 * (i >> 3) + 8 * half + (i & 7)]);
            bf1.u[i] = f32_to_bf16_rne(kp[32 + 16 * (i >> 3) + 8 * half + (i & 7)]);
        }
    }

    const unsigned long long qbase =
        (unsigned long long)(const void*)(Q + ((size_t)b * Sq * Hh + h) * (size_t)Ll);
    const unsigned ldsFaddr = lds_offset(ldsF);

    // Prologue: DMA tile 0 (wave 0 owns the TDM queue for this block)
    if (wave == 0)
        tdm_load_qtile(ldsFaddr, qbase);

    float runM = -__builtin_inff();
    float runZ = 0.0f;
    const size_t maskBase = (size_t)b * Sq * Sq + (size_t)(k0 + n);

    for (int q0 = 0; q0 < Sq; q0 += 16) {
        // Close the pipeline stage: DMA of tile q0 done, prior WMMA reads done.
        __builtin_amdgcn_s_wait_tensorcnt(0);
        __syncthreads();

        // Convert the raw f32 tile to bf16 in the WMMA staging buffer.
        for (int t = tid; t < 16 * Ll; t += blockDim.x)
            qtile[t >> 6][t & 63] = f32_to_bf16_rne(ldsF[t]);
        __syncthreads();

        // Overlap: kick the DMA for tile q0+16 while this tile computes.
        if (wave == 0 && (q0 + 16) < Sq)
            tdm_load_qtile(ldsFaddr,
                           qbase + (unsigned long long)(q0 + 16) * Hh * Ll * 4ull);

        // A fragments from LDS (16B-aligned b128 DS loads, ISA A-layout):
        // lane -> row m = lane%16; elems 0-7 -> l = 8*half+i ; 8-15 -> 16+8*half+i-8
        const int m = lane & 15;
        FragBF16 a0, a1;
        a0.q[0] = *reinterpret_cast<const uint4*>(&qtile[m][ 8 * half]);
        a0.q[1] = *reinterpret_cast<const uint4*>(&qtile[m][16 + 8 * half]);
        a1.q[0] = *reinterpret_cast<const uint4*>(&qtile[m][32 + 8 * half]);
        a1.q[1] = *reinterpret_cast<const uint4*>(&qtile[m][48 + 8 * half]);

        v8f c = {};
        c = __builtin_amdgcn_wmma_f32_16x16x32_bf16(false, a0.v, false, bf0.v,
                                                    (short)0, c, false, false);
        c = __builtin_amdgcn_wmma_f32_16x16x32_bf16(false, a1.v, false, bf1.v,
                                                    (short)0, c, false, false);

        // Prefetch next q-tile's mask rows (global_prefetch_b8)
        if (q0 + 16 < Sq)
            __builtin_prefetch(&mask[maskBase + (size_t)(q0 + 16 + 8 * half) * Sq], 0, 0);

        // C layout: lane column n, VGPR r -> row q0 + r + 8*half.
        // Mask, scale by 1/sqrt(64), online softmax along q for this half-column.
        float sv[8];
        float tileMax = -__builtin_inff();
#pragma unroll
        for (int r = 0; r < 8; ++r) {
            const int qrow = q0 + r + 8 * half;
            const int mk = mask[maskBase + (size_t)qrow * Sq];
            float s = c[r] * 0.125f;
            s = (mk == 0) ? -1.0e10f : s;
            sv[r] = s;
            tileMax = fmaxf(tileMax, s);
        }
        const float newM = fmaxf(runM, tileMax);
        float acc = 0.0f;
#pragma unroll
        for (int r = 0; r < 8; ++r) acc += __expf(sv[r] - newM);
        runZ = runZ * __expf(runM - newM) + acc;
        runM = newM;
    }

    // Merge the two row-halves of each column (lanes t <-> t^16 share column n)
    const float oM = __shfl_xor(runM, 16, 32);
    const float oZ = __shfl_xor(runZ, 16, 32);
    const float M  = fmaxf(runM, oM);
    const float Z  = runZ * __expf(runM - M) + oZ * __expf(oM - M);

    // Faithful per-column normalized sum: sum_q exp(s-M)/Z == Z/Z
    float contrib = Z / Z;
    if (half != 0) contrib = 0.0f;          // count each column once
#pragma unroll
    for (int off = 16; off >= 1; off >>= 1) contrib += __shfl_xor(contrib, off, 32);
    if (lane == 0) atomicAdd(&Sbh[b * Hh + h], contrib);
}

// ---------------------------------------------------------------------------
// Kernel 2: out[b,v,h,l] = values[b,v,h,l] * S[b,h]   (bandwidth-bound, f4)
// ---------------------------------------------------------------------------
__global__ void scale_values_kernel(const float* __restrict__ V,
                                    const float* __restrict__ Sbh,
                                    float*       __restrict__ out,
                                    int n4) {
    const int i = blockIdx.x * blockDim.x + threadIdx.x;
    if (i >= n4) return;
    const int flat = i << 2;                 // 4 consecutive l share (b,h)
    const int h = (flat >> 6) & 15;          // L = 64
    const int b = flat >> 21;                // S*H*L = 2^21
    const float s = Sbh[b * Hh + h];
    float4 v = reinterpret_cast<const float4*>(V)[i];
    v.x *= s; v.y *= s; v.z *= s; v.w *= s;
    reinterpret_cast<float4*>(out)[i] = v;
}

// ---------------------------------------------------------------------------
extern "C" void kernel_launch(void* const* d_in, const int* in_sizes, int n_in,
                              void* d_out, int out_size, void* d_ws, size_t ws_size,
                              hipStream_t stream) {
    const float* Q    = (const float*)d_in[0];
    const float* K    = (const float*)d_in[1];
    const float* V    = (const float*)d_in[2];
    const int*   mask = (const int*)d_in[3];
    float* out = (float*)d_out;
    float* Sbh = (float*)d_ws;               // 64 floats of scratch

    zero_sbh_kernel<<<1, 64, 0, stream>>>(Sbh, Bq * Hh);

    // B*H*(S/64) = 4*16*32 = 2048 blocks, 128 threads (4 wave32)
    attn_colsum_kernel<<<2048, 128, 0, stream>>>(Q, K, mask, Sbh);

    const int n4 = (Bq * Sq * Hh * Ll) / 4;  // 2,097,152 float4
    scale_values_kernel<<<(n4 + 255) / 256, 256, 0, stream>>>(V, Sbh, out, n4);
}
